// DeformableConvBlock_70514773065939
// MI455X (gfx1250) — matrix operand order repaired and checked by reading, not compiled, not run on hardware
//
#include <hip/hip_runtime.h>

// ---------------------------------------------------------------------------
// DeformableConvBlock for MI455X (gfx1250), wave32, WMMA f16->f32.
// B=4, C=128, H=W=96.
//   offset = conv3x3(concat(lr,hr))        GEMM [32pad x 2304] x [2304 x N]
//   out    = deform_conv3x3(hr, offset)    GEMM [128 x 1152] x [1152 x N]
// Offset GEMM K-order: k = tap*256 + c   (chunk of 32 never crosses a tap)
// Deform GEMM K-order: k = c*9 + tap
// ---------------------------------------------------------------------------

typedef __attribute__((ext_vector_type(16))) _Float16 v16h;
typedef __attribute__((ext_vector_type(8)))  _Float16 v8h;
typedef __attribute__((ext_vector_type(8)))  float    v8f;

#define Bb 4
#define Cc 128
#define Hh 96
#define Ww 96

// ---------------- A-fragment prepack ---------------------------------------
// 16-bit A (16x32 MxK) per-lane K order chosen here (contiguous; B uses the same): K(e,lane) = (lane>=16)*16 + e
// Packed so a wave loads its fragment as 16 contiguous _Float16 per lane.

__global__ void pack_w_off_kernel(const float* __restrict__ w_off,
                                  _Float16* __restrict__ A) {
    int idx = blockIdx.x * 256 + threadIdx.x;          // 2*72*32*16 = 73728
    if (idx >= 2 * 72 * 512) return;
    int e    = idx & 15;
    int lane = (idx >> 4) & 31;
    int q    = (idx >> 9) % 72;
    int mt   = idx / (72 * 512);
    int m    = mt * 16 + (lane & 15);
    int k    = q * 32 + (lane >> 4) * 16 + e;   // < 2304
    int tap  = k >> 8;                                  // k = tap*256 + c
    int c    = k & 255;
    float v  = (m < 18) ? w_off[(m * 256 + c) * 9 + tap] : 0.0f;
    A[idx] = (_Float16)v;
}

__global__ void pack_w_def_kernel(const float* __restrict__ w_def,
                                  _Float16* __restrict__ A) {
    int idx = blockIdx.x * 256 + threadIdx.x;          // 8*36*32*16 = 147456
    if (idx >= 8 * 36 * 512) return;
    int e    = idx & 15;
    int lane = (idx >> 4) & 31;
    int q    = (idx >> 9) % 36;
    int mt   = idx / (36 * 512);
    int m    = mt * 16 + (lane & 15);                   // < 128
    int k    = q * 32 + (lane >> 4) * 16 + e;   // < 1152
    int c    = k / 9;                                   // k = c*9 + tap
    int tap  = k % 9;
    A[idx] = (_Float16)w_def[(m * 128 + c) * 9 + tap];
}

// ---------------- offset conv: 18ch 3x3 over concat(lr,hr) -----------------
// Block = 128 threads (4 waves). Tile = one row, 32 pixels.
// LDS patch (transposed): [3 rows][34 cols][256 ch] f16 = 52224 B, channel
// fastest so a B-fragment (16 consecutive channels of one tap/pixel) is two
// ds_load_b128. 4 waves = 2 M-tiles (18->32 pad) x 2 N-tiles (16 px).

__global__ __launch_bounds__(128) void offset_conv_kernel(
        const float* __restrict__ lr, const float* __restrict__ hr,
        const _Float16* __restrict__ A_off, const float* __restrict__ b_off,
        float* __restrict__ off_out) {
    __shared__ _Float16 patch_t[3 * 34 * 256];   // [(r*34+col)*256 + c]

    int tile = blockIdx.x;                 // 0 .. 4*96*3-1
    int b    = tile / (Hh * 3);
    int rem  = tile % (Hh * 3);
    int y    = rem / 3;
    int x0   = (rem % 3) * 32;
    int tid  = threadIdx.x;

    // fill: iterate with col fastest for coalesced global reads; scatter to LDS
    for (int idx = tid; idx < 256 * 3 * 34; idx += 128) {
        int c   = idx / 102;
        int r   = (idx % 102) / 34;
        int col = idx % 34;
        int gy  = y + r - 1;
        int gx  = x0 + col - 1;
        float v = 0.0f;
        if (gy >= 0 && gy < Hh && gx >= 0 && gx < Ww) {
            const float* src = (c < 128) ? lr : hr;
            int cc = (c < 128) ? c : (c - 128);
            v = src[((b * 128 + cc) * Hh + gy) * Ww + gx];
        }
        patch_t[(r * 34 + col) * 256 + c] = (_Float16)v;
    }
    __syncthreads();

    int wv   = tid >> 5;                   // 0..3
    int lane = tid & 31;
    int mt   = wv >> 1;                    // output-channel tile (0..1)
    int nt   = wv & 1;                     // pixel tile (0..1)
    int n    = lane & 15;
    int lh   = lane >> 4;

    v8f acc = {};
    #pragma unroll 2
    for (int q = 0; q < 72; ++q) {
        v16h a = *(const v16h*)(A_off + (((mt * 72 + q) * 32 + lane) << 4));
        int tap = q >> 3;                  // uniform per iteration
        int ky  = tap / 3;
        int kx  = tap % 3;
        int cb  = (q & 7) * 32 + lh * 16;  // 16 consecutive channels
        const _Float16* bp =
            &patch_t[(ky * 34 + (nt * 16 + n + kx)) * 256 + cb];
        v8h lo = *(const v8h*)(bp);
        v8h hi = *(const v8h*)(bp + 8);
        v16h bm = __builtin_shufflevector(lo, hi,
                      0, 1, 2, 3, 4, 5, 6, 7, 8, 9, 10, 11, 12, 13, 14, 15);
        acc = __builtin_amdgcn_wmma_f32_16x16x32_f16(
                  false, a, false, bm, (short)0, acc, false, false);
    }

    #pragma unroll
    for (int r = 0; r < 8; ++r) {
        int m  = r + lh * 8;
        int co = mt * 16 + m;
        if (co < 18) {
            off_out[((b * 18 + co) * Hh + y) * Ww + (x0 + nt * 16 + n)] =
                acc[r] + b_off[co];
        }
    }
}

// ---------------- deformable conv ------------------------------------------
// Block = 256 threads (8 waves). Tile = one row, 16 pixels, all 128 out ch.
// Phase 1: bilinear coefficients for 9 taps x 16 pixels.
// Phase 2: gather 128ch x 9tap x 16px samples from hr (L2 resident) into an
//          LDS im2col panel samp[pixel][k], k = c*9+tap, padded stride.
// Phase 3: wave w = M-tile w; 36 x v_wmma_f32_16x16x32_f16.

#define KSTRIDE 1160   // 1152 + 8 pad; 1160*2 bytes keeps rows 16B aligned

__global__ __launch_bounds__(256) void deform_conv_kernel(
        const float* __restrict__ hr, const float* __restrict__ off_buf,
        const _Float16* __restrict__ A_def, const float* __restrict__ b_def,
        float* __restrict__ out) {
    __shared__ int      s_y0[144];
    __shared__ int      s_x0[144];
    __shared__ float    s_w[144][4];
    __shared__ _Float16 samp[16 * KSTRIDE];

    int tile = blockIdx.x;                 // 0 .. 4*96*6-1
    int b    = tile / (Hh * 6);
    int rem  = tile % (Hh * 6);
    int y    = rem / 6;
    int x0   = (rem % 6) * 16;
    int tid  = threadIdx.x;

    if (tid < 144) {
        int tap = tid >> 4;
        int pix = tid & 15;
        int x   = x0 + pix;
        float dy = off_buf[((b * 18 + tap * 2    ) * Hh + y) * Ww + x];
        float dx = off_buf[((b * 18 + tap * 2 + 1) * Hh + y) * Ww + x];
        float sy = dy + (float)(tap / 3 - 1) + (float)y;
        float sx = dx + (float)(tap % 3 - 1) + (float)x;
        float y0f = floorf(sy);
        float x0f = floorf(sx);
        float fy  = sy - y0f;
        float fx  = sx - x0f;
        s_y0[tid] = (int)y0f;
        s_x0[tid] = (int)x0f;
        s_w[tid][0] = (1.0f - fy) * (1.0f - fx);
        s_w[tid][1] = (1.0f - fy) * fx;
        s_w[tid][2] = fy * (1.0f - fx);
        s_w[tid][3] = fy * fx;
    }
    __syncthreads();

    for (int s = tid; s < 128 * 144; s += 256) {
        int c   = s / 144;
        int pt  = s % 144;
        int tap = pt >> 4;
        int pix = pt & 15;
        int iy  = s_y0[pt];
        int ix  = s_x0[pt];
        const float* base = hr + (size_t)(b * 128 + c) * (Hh * Ww);
        float v00 = 0.0f, v01 = 0.0f, v10 = 0.0f, v11 = 0.0f;
        bool y0ok = (iy >= 0)     && (iy < Hh);
        bool y1ok = (iy + 1 >= 0) && (iy + 1 < Hh);
        bool x0ok = (ix >= 0)     && (ix < Ww);
        bool x1ok = (ix + 1 >= 0) && (ix + 1 < Ww);
        if (y0ok && x0ok) v00 = base[iy * Ww + ix];
        if (y0ok && x1ok) v01 = base[iy * Ww + ix + 1];
        if (y1ok && x0ok) v10 = base[(iy + 1) * Ww + ix];
        if (y1ok && x1ok) v11 = base[(iy + 1) * Ww + ix + 1];
        float val = v00 * s_w[pt][0] + v01 * s_w[pt][1] +
                    v10 * s_w[pt][2] + v11 * s_w[pt][3];
        samp[pix * KSTRIDE + (c * 9 + tap)] = (_Float16)val;
    }
    __syncthreads();

    int wv   = tid >> 5;                   // M-tile 0..7 -> out ch 16*wv..
    int lane = tid & 31;
    int n    = lane & 15;                  // pixel
    int lh   = lane >> 4;

    v8f acc = {};
    #pragma unroll 2
    for (int q = 0; q < 36; ++q) {
        v16h a = *(const v16h*)(A_def + (((wv * 36 + q) * 32 + lane) << 4));
        int kb = q * 32 + lh * 16;
        const _Float16* bp = &samp[n * KSTRIDE + kb];
        v8h lo = *(const v8h*)(bp);
        v8h hi = *(const v8h*)(bp + 8);
        v16h bm = __builtin_shufflevector(lo, hi,
                      0, 1, 2, 3, 4, 5, 6, 7, 8, 9, 10, 11, 12, 13, 14, 15);
        acc = __builtin_amdgcn_wmma_f32_16x16x32_f16(
                  false, a, false, bm, (short)0, acc, false, false);
    }

    #pragma unroll
    for (int r = 0; r < 8; ++r) {
        int m  = r + lh * 8;
        int co = wv * 16 + m;
        out[((size_t)(b * 128 + co) * Hh + y) * Ww + (x0 + n)] =
            acc[r] + b_def[co];
    }
}

// ---------------------------------------------------------------------------
extern "C" void kernel_launch(void* const* d_in, const int* in_sizes, int n_in,
                              void* d_out, int out_size, void* d_ws, size_t ws_size,
                              hipStream_t stream) {
    const float* lr    = (const float*)d_in[0];   // [4,128,96,96]
    const float* hr    = (const float*)d_in[1];   // [4,128,96,96]
    const float* w_off = (const float*)d_in[2];   // [18,256,3,3]
    const float* b_off = (const float*)d_in[3];   // [18]
    const float* w_def = (const float*)d_in[4];   // [128,128,3,3]
    const float* b_def = (const float*)d_in[5];   // [128]
    float* out = (float*)d_out;                   // [4,128,96,96]

    // workspace layout (~3.1 MB): offsets fp32, then prepacked f16 A-panels
    const size_t off_elems = (size_t)Bb * 18 * Hh * Ww;           // 663552
    float*    off_buf = (float*)d_ws;
    _Float16* A_off   = (_Float16*)((char*)d_ws + off_elems * sizeof(float));
    _Float16* A_def   = A_off + 2 * 72 * 512;                     // 73728 f16

    pack_w_off_kernel<<<288, 256, 0, stream>>>(w_off, A_off);
    pack_w_def_kernel<<<576, 256, 0, stream>>>(w_def, A_def);
    offset_conv_kernel<<<Bb * Hh * 3, 128, 0, stream>>>(lr, hr, A_off, b_off, off_buf);
    deform_conv_kernel<<<Bb * Hh * 6, 256, 0, stream>>>(hr, off_buf, A_def, b_def, out);
    (void)in_sizes; (void)n_in; (void)out_size; (void)ws_size;
}
